// VQVAE_70403103916771
// MI455X (gfx1250) — matrix operand order repaired and checked by reading, not compiled, not run on hardware
//
#include <hip/hip_runtime.h>
#include <cstddef>

// ---------- vector types for WMMA ----------
typedef __bf16  bf16x8  __attribute__((ext_vector_type(8)));
typedef __bf16  bf16x16 __attribute__((ext_vector_type(16)));
typedef float   f32x8   __attribute__((ext_vector_type(8)));

#define CAT16(lo, hi) __builtin_shufflevector((lo), (hi), 0,1,2,3,4,5,6,7,8,9,10,11,12,13,14,15)

// Load one 16x32 bf16 A/B-style fragment from an LDS tile stored [row][32]
__device__ __forceinline__ bf16x16 frag_a(const __bf16* base, int row, int kh)
{
    const __bf16* p = base + row * 32 + kh * 8;
    bf16x8 lo = *(const bf16x8*)p;          // K = kh*8 .. +7
    bf16x8 hi = *(const bf16x8*)(p + 16);   // K = 16+kh*8 .. +7
    return CAT16(lo, hi);
}
__device__ __forceinline__ bf16x16 frag_b(const __bf16* base, int row, int kh)
{
    const __bf16* p = base + row * 32 + kh * 8;
    bf16x8 lo = *(const bf16x8*)p;          // K = kh*8 .. +7
    bf16x8 hi = *(const bf16x8*)(p + 16);   // K = 16+kh*8 .. +7
    return CAT16(lo, hi);
}

// =====================================================================
// Implicit-GEMM convolution (KS=3 SAME or KS=1), bf16 WMMA, f32 accum.
// Block: 256 threads = 8 waves; block tile M=64 (Co) x N=128 (pixels).
// Wave tile: 32x32 -> 4 accumulators, 2 A-frags x 2 B-frags reused into
// 4 back-to-back v_wmma per K-chunk per wave between barriers.
// ACT: 0=none, 1=relu, 2=tanh
// =====================================================================
template <int KS, int ACT>
__global__ void __launch_bounds__(256)
conv_wmma_k(const float* __restrict__ in, const float* __restrict__ wgt,
            const float* __restrict__ bias, float* __restrict__ out,
            int Ci, int Co, int H, int W)
{
    __shared__ __bf16 Ab[64 * 32];    // weights  [m][k]   4 KB
    __shared__ __bf16 Bb[128 * 32];   // im2col   [n][k]   8 KB

    const int t    = threadIdx.x;
    const int lane = t & 31;
    const int wave = t >> 5;
    const int co0  = blockIdx.x * 64;
    const int p0   = blockIdx.y * 128;
    const int nimg = blockIdx.z;

    const size_t HW  = (size_t)H * W;
    const float* inN = in + (size_t)nimg * Ci * HW;

    // B-stage coords: this thread fills Bb[bn][bk16..bk16+15]
    // lanes run over consecutive pixels -> coalesced 128B global loads
    const int bn   = t & 127;
    const int bk16 = (t >> 7) * 16;
    const int p    = p0 + bn;
    const int ph   = p / W;
    const int pw   = p % W;

    // A-stage coords: this thread fills Ab[am][ak8..ak8+7]
    const int am  = t >> 2;
    const int ak8 = (t & 3) * 8;

    const int mr = lane & 15;   // row within 16x16 tile
    const int kh = lane >> 4;   // lane half

    const int mw = wave & 1;    // 0..1 -> 32-channel half
    const int nw = wave >> 1;   // 0..3 -> 32-pixel quarter

    f32x8 acc00 = {}, acc01 = {}, acc10 = {}, acc11 = {};

    for (int ky = 0; ky < KS; ++ky) {
        const int ih = ph + ky - KS / 2;
        for (int kx = 0; kx < KS; ++kx) {
            const int  iw  = pw + kx - KS / 2;
            const bool inb = (ih >= 0) && (ih < H) && (iw >= 0) && (iw < W);
            const float* src = inN + (size_t)ih * W + iw;

            for (int c0 = 0; c0 < Ci; c0 += 32) {
                __syncthreads();
                // ---- stage B tile (activations, im2col) [128 px][32 k] ----
                #pragma unroll
                for (int j = 0; j < 16; ++j) {
                    const int k  = bk16 + j;
                    const int ci = c0 + k;
                    float v = (inb && ci < Ci) ? src[(size_t)ci * HW] : 0.0f;
                    Bb[bn * 32 + k] = (__bf16)v;
                }
                if (inb && (c0 + 32) < Ci)
                    __builtin_prefetch(src + (size_t)(c0 + 32) * HW, 0, 3); // global_prefetch_b8
                // ---- stage A tile (weights) [64 co][32 k] ----
                {
                    const int co = co0 + am;
                    #pragma unroll
                    for (int j = 0; j < 8; ++j) {
                        const int k  = ak8 + j;
                        const int ci = c0 + k;
                        float v = (co < Co && ci < Ci)
                                ? wgt[(((size_t)co * Ci + ci) * KS + ky) * KS + kx]
                                : 0.0f;
                        Ab[am * 32 + k] = (__bf16)v;
                    }
                }
                __syncthreads();

                // ---- fragments (ISA 7.12.2 layouts) + 4x WMMA ----
                bf16x16 af0 = frag_a(Ab, mw * 32 + mr,      kh);
                bf16x16 af1 = frag_a(Ab, mw * 32 + 16 + mr, kh);
                bf16x16 bf0 = frag_b(Bb, nw * 32 + mr,      kh);
                bf16x16 bf1 = frag_b(Bb, nw * 32 + 16 + mr, kh);

                acc00 = __builtin_amdgcn_wmma_f32_16x16x32_bf16(
                        false, af0, false, bf0, (short)0, acc00, false, false);
                acc01 = __builtin_amdgcn_wmma_f32_16x16x32_bf16(
                        false, af0, false, bf1, (short)0, acc01, false, false);
                acc10 = __builtin_amdgcn_wmma_f32_16x16x32_bf16(
                        false, af1, false, bf0, (short)0, acc10, false, false);
                acc11 = __builtin_amdgcn_wmma_f32_16x16x32_bf16(
                        false, af1, false, bf1, (short)0, acc11, false, false);
            }
        }
    }

    // ---- store D tiles: VGPR r <-> row r + 8*kh, col = lane&15 ----
    const int pp0 = p0 + nw * 32 + mr;
    const int pp1 = pp0 + 16;
    #pragma unroll
    for (int r = 0; r < 8; ++r) {
        const int coA = co0 + mw * 32 + kh * 8 + r;   // rows of af0 tile
        const int coB = coA + 16;                     // rows of af1 tile
        if (coA < Co) {
            const float b = bias[coA];
            float v0 = acc00[r] + b;
            float v1 = acc01[r] + b;
            if (ACT == 1) { v0 = fmaxf(v0, 0.0f); v1 = fmaxf(v1, 0.0f); }
            if (ACT == 2) { v0 = tanhf(v0);       v1 = tanhf(v1); }
            float* o = out + ((size_t)nimg * Co + coA) * HW;
            o[pp0] = v0;
            o[pp1] = v1;
        }
        if (coB < Co) {
            const float b = bias[coB];
            float v0 = acc10[r] + b;
            float v1 = acc11[r] + b;
            if (ACT == 1) { v0 = fmaxf(v0, 0.0f); v1 = fmaxf(v1, 0.0f); }
            if (ACT == 2) { v0 = tanhf(v0);       v1 = tanhf(v1); }
            float* o = out + ((size_t)nimg * Co + coB) * HW;
            o[pp0] = v0;
            o[pp1] = v1;
        }
    }
}

// =====================================================================
// Conv-transpose 2x2 stride 2 (VALID): one (dy,dx) slice == 1x1 GEMM,
// output written to disjoint pixels (2h+dy, 2w+dx). Weight [co][ci][2][2].
// Same M=64 x N=128 tiling, 4 WMMAs per wave per chunk.
// =====================================================================
template <int ACT>
__global__ void __launch_bounds__(256)
convt_wmma_k(const float* __restrict__ in, const float* __restrict__ wgt,
             const float* __restrict__ bias, float* __restrict__ out,
             int Ci, int Co, int H, int W, int dy, int dx)
{
    __shared__ __bf16 Ab[64 * 32];
    __shared__ __bf16 Bb[128 * 32];

    const int t    = threadIdx.x;
    const int lane = t & 31;
    const int wave = t >> 5;
    const int co0  = blockIdx.x * 64;
    const int p0   = blockIdx.y * 128;
    const int nimg = blockIdx.z;

    const size_t HW  = (size_t)H * W;
    const float* inN = in + (size_t)nimg * Ci * HW;

    const int bn   = t & 127;
    const int bk16 = (t >> 7) * 16;
    const int p    = p0 + bn;

    const int am  = t >> 2;
    const int ak8 = (t & 3) * 8;

    const int mr = lane & 15;
    const int kh = lane >> 4;
    const int mw = wave & 1;
    const int nw = wave >> 1;

    f32x8 acc00 = {}, acc01 = {}, acc10 = {}, acc11 = {};
    const float* src = inN + p;

    for (int c0 = 0; c0 < Ci; c0 += 32) {
        __syncthreads();
        #pragma unroll
        for (int j = 0; j < 16; ++j) {
            const int k  = bk16 + j;
            const int ci = c0 + k;
            float v = (ci < Ci) ? src[(size_t)ci * HW] : 0.0f;
            Bb[bn * 32 + k] = (__bf16)v;
        }
        if ((c0 + 32) < Ci)
            __builtin_prefetch(src + (size_t)(c0 + 32) * HW, 0, 3);
        {
            const int co = co0 + am;
            #pragma unroll
            for (int j = 0; j < 8; ++j) {
                const int k  = ak8 + j;
                const int ci = c0 + k;
                float v = (co < Co && ci < Ci)
                        ? wgt[(((size_t)co * Ci + ci) * 2 + dy) * 2 + dx]
                        : 0.0f;
                Ab[am * 32 + k] = (__bf16)v;
            }
        }
        __syncthreads();

        bf16x16 af0 = frag_a(Ab, mw * 32 + mr,      kh);
        bf16x16 af1 = frag_a(Ab, mw * 32 + 16 + mr, kh);
        bf16x16 bf0 = frag_b(Bb, nw * 32 + mr,      kh);
        bf16x16 bf1 = frag_b(Bb, nw * 32 + 16 + mr, kh);

        acc00 = __builtin_amdgcn_wmma_f32_16x16x32_bf16(
                false, af0, false, bf0, (short)0, acc00, false, false);
        acc01 = __builtin_amdgcn_wmma_f32_16x16x32_bf16(
                false, af0, false, bf1, (short)0, acc01, false, false);
        acc10 = __builtin_amdgcn_wmma_f32_16x16x32_bf16(
                false, af1, false, bf0, (short)0, acc10, false, false);
        acc11 = __builtin_amdgcn_wmma_f32_16x16x32_bf16(
                false, af1, false, bf1, (short)0, acc11, false, false);
    }

    const int pq0 = p0 + nw * 32 + mr;
    const int pq1 = pq0 + 16;
    const int oh0 = pq0 / W, ow0 = pq0 % W;
    const int oh1 = pq1 / W, ow1 = pq1 % W;
    const size_t oHW = HW * 4;
    const size_t od0 = (size_t)(2 * oh0 + dy) * (2 * W) + (2 * ow0 + dx);
    const size_t od1 = (size_t)(2 * oh1 + dy) * (2 * W) + (2 * ow1 + dx);
    #pragma unroll
    for (int r = 0; r < 8; ++r) {
        const int coA = co0 + mw * 32 + kh * 8 + r;
        const int coB = coA + 16;
        if (coA < Co) {
            const float b = bias[coA];
            float v0 = acc00[r] + b;
            float v1 = acc01[r] + b;
            if (ACT == 2) { v0 = tanhf(v0); v1 = tanhf(v1); }
            float* o = out + ((size_t)nimg * Co + coA) * oHW;
            o[od0] = v0;
            o[od1] = v1;
        }
        if (coB < Co) {
            const float b = bias[coB];
            float v0 = acc10[r] + b;
            float v1 = acc11[r] + b;
            if (ACT == 2) { v0 = tanhf(v0); v1 = tanhf(v1); }
            float* o = out + ((size_t)nimg * Co + coB) * oHW;
            o[od0] = v0;
            o[od1] = v1;
        }
    }
}

// =====================================================================
// MaxPool 2x2 stride 2
// =====================================================================
__global__ void maxpool2_k(const float* __restrict__ in, float* __restrict__ out,
                           int C, int H)
{
    const int Ho = H >> 1;
    const size_t total = (size_t)4 * C * Ho * Ho;
    const size_t i = (size_t)blockIdx.x * 256 + threadIdx.x;
    if (i >= total) return;
    const int w = (int)(i % Ho);
    const int h = (int)((i / Ho) % Ho);
    const int c = (int)((i / ((size_t)Ho * Ho)) % C);
    const int n = (int)(i / ((size_t)Ho * Ho * C));
    const float* p = in + ((size_t)(n * C + c) * H + 2 * h) * H + 2 * w;
    out[i] = fmaxf(fmaxf(p[0], p[1]), fmaxf(p[H], p[H + 1]));
}

// =====================================================================
// VQ: ||e||^2 precompute
// =====================================================================
__global__ void emb_norm_k(const float* __restrict__ emb, float* __restrict__ enorm)
{
    const int e = blockIdx.x * 256 + threadIdx.x;
    if (e < 1024) {
        float s = 0.0f;
        #pragma unroll 8
        for (int c = 0; c < 64; ++c) { float v = emb[e * 64 + c]; s += v * v; }
        enorm[e] = s;
    }
}

// =====================================================================
// VQ argmin: per block 16 rows of z_e_flat vs all 1024 embeddings.
// d = ||e||^2 - 2 z.e  (||z||^2 constant per row). WMMA for z.e^T.
// =====================================================================
__global__ void __launch_bounds__(256)
vq_argmin_k(const float* __restrict__ ze, const float* __restrict__ emb,
            const float* __restrict__ enorm, int* __restrict__ outidx)
{
    __shared__ __bf16 Zb[16 * 64];
    __shared__ float  RV[2048];
    __shared__ int    RI[2048];

    const int t    = threadIdx.x;
    const int lane = t & 31;
    const int wave = t >> 5;
    const int r0   = blockIdx.x * 16;

    // load 16 rows (row = b*256 + h*16 + w), all 64 channels, NCHW source
    {
        const int i  = t >> 4;
        const int cb = (t & 15) * 4;
        const int n  = r0 + i;
        const int b  = n >> 8;
        const int hw = n & 255;
        #pragma unroll
        for (int j = 0; j < 4; ++j) {
            const int c = cb + j;
            Zb[i * 64 + c] = (__bf16)ze[(((size_t)b * 64 + c) << 8) + hw];
        }
    }
    __syncthreads();

    float bestv[8];
    int   besti[8];
    #pragma unroll
    for (int r = 0; r < 8; ++r) { bestv[r] = 3.4e38f; besti[r] = 0; }

    const int mr = lane & 15;
    const int kh = lane >> 4;

    for (int tt = 0; tt < 8; ++tt) {
        const int e0 = tt * 128 + wave * 16;
        f32x8 acc = {};
        #pragma unroll
        for (int c0 = 0; c0 < 64; c0 += 32) {
            bf16x16 af = frag_a(Zb + c0, mr, kh);   // rows are 64 wide: adjust below
            // NOTE: Zb stride is 64, not 32 -> load manually
            {
                const __bf16* ap = &Zb[mr * 64 + c0 + kh * 8];
                bf16x8 alo = *(const bf16x8*)ap;
                bf16x8 ahi = *(const bf16x8*)(ap + 16);
                af = CAT16(alo, ahi);
            }

            // B fragment straight from global: emb row e0+mr, C = kh*8..+7 and 16+kh*8..+7
            const float* bp = emb + (size_t)(e0 + mr) * 64 + c0 + kh * 8;
            bf16x16 bfv;
            #pragma unroll
            for (int j = 0; j < 16; ++j) bfv[j] = (__bf16)bp[j + (j >= 8 ? 8 : 0)];

            acc = __builtin_amdgcn_wmma_f32_16x16x32_bf16(
                    false, af, false, bfv, (short)0, acc, false, false);
        }
        const int   e  = e0 + mr;
        const float en = enorm[e];
        #pragma unroll
        for (int r = 0; r < 8; ++r) {
            const float d = en - 2.0f * acc[r];
            if (d < bestv[r]) { bestv[r] = d; besti[r] = e; }
        }
    }

    #pragma unroll
    for (int r = 0; r < 8; ++r) { RV[(t << 3) + r] = bestv[r]; RI[(t << 3) + r] = besti[r]; }
    __syncthreads();

    if (t < 16) {
        const int row = t;
        const int khr = row >> 3;
        const int rr  = row & 7;
        float bv = 3.4e38f; int bi = 0;
        for (int w = 0; w < 8; ++w)
            for (int j = 0; j < 16; ++j) {
                const int l  = khr * 16 + j;
                const int id = ((w * 32 + l) << 3) + rr;
                const float v = RV[id];
                if (v < bv) { bv = v; bi = RI[id]; }
            }
        outidx[r0 + row] = bi;
    }
}

// =====================================================================
// VQ gather + loss accumulation (forward commit==codebook numerically)
// =====================================================================
__global__ void vq_gather_k(const int* __restrict__ idx, const float* __restrict__ emb,
                            const float* __restrict__ ze, float* __restrict__ latent,
                            float* __restrict__ lsum)
{
    const int n = blockIdx.x * 256 + threadIdx.x;
    if (n >= 1024) return;
    const int b  = n >> 8;
    const int hw = n & 255;
    const int e  = idx[n];
    float s = 0.0f;
    for (int c = 0; c < 64; ++c) {
        const size_t off = (((size_t)b * 64 + c) << 8) + hw;
        const float zq = emb[e * 64 + c];
        const float d  = zq - ze[off];
        s += d * d;
        latent[off] = zq;   // straight-through forward value = z_q
    }
    atomicAdd(lsum, s);
}

__global__ void zero_k(float* p) { if (threadIdx.x == 0 && blockIdx.x == 0) p[0] = 0.0f; }

__global__ void qloss_k(const float* __restrict__ lsum, float* __restrict__ q)
{
    // q_loss = codebook + 0.25*commitment = 1.25 * mean((zq-ze)^2) over 1024*64
    if (threadIdx.x == 0 && blockIdx.x == 0) q[0] = lsum[0] * (1.25f / 65536.0f);
}

// =====================================================================
// Host sequencing
// =====================================================================
extern "C" void kernel_launch(void* const* d_in, const int* in_sizes, int n_in,
                              void* d_out, int out_size, void* d_ws, size_t ws_size,
                              hipStream_t stream)
{
    (void)in_sizes; (void)n_in; (void)out_size; (void)ws_size;

    // ---- JAX pytree (sorted-key) flatten order of setup_inputs() ----
    // params.dec_blk[4 x (w1,b1,w2,b2)]: 0..15
    // params.dec_ct [5 x (w,b)]        : 16..25
    // params.emb                       : 26
    // params.enc    [5 x (w1,b1,w2,b2)]: 27..46
    // params.post_b:47  post_w:48  pre_b:49  pre_w:50   x:51
    auto fin = [&](int i) { return (const float*)d_in[i]; };

    float* bufA = (float*)d_ws;
    float* bufB = (float*)((char*)d_ws + (size_t)268435456);          // 64M f32
    float* smal = (float*)((char*)d_ws + (size_t)536870912);
    float* enorm = smal;                 // 1024 f32
    int*   vidx  = (int*)(smal + 1024);  // 1024 i32
    float* lsum  = smal + 2048;          // 1 f32

    float* recon  = (float*)d_out;       // 4*3*512*512
    float* latent = recon + 3145728;     // 4*64*16*16
    float* qloss  = latent + 65536;      // 1

    auto conv3 = [&](const float* src, int wi, int bi, float* dst,
                     int Ci, int Co, int H, bool relu) {
        dim3 g((unsigned)((Co + 63) / 64), (unsigned)((H * H) / 128), 4u);
        if (relu) conv_wmma_k<3, 1><<<g, 256, 0, stream>>>(src, fin(wi), fin(bi), dst, Ci, Co, H, H);
        else      conv_wmma_k<3, 0><<<g, 256, 0, stream>>>(src, fin(wi), fin(bi), dst, Ci, Co, H, H);
    };
    auto conv1 = [&](const float* src, int wi, int bi, float* dst,
                     int Ci, int Co, int H) {
        dim3 g((unsigned)((Co + 63) / 64), (unsigned)((H * H) / 128), 4u);
        conv_wmma_k<1, 0><<<g, 256, 0, stream>>>(src, fin(wi), fin(bi), dst, Ci, Co, H, H);
    };
    auto pool = [&](const float* src, float* dst, int C, int H) {
        size_t total = (size_t)4 * C * (H / 2) * (H / 2);
        maxpool2_k<<<(unsigned)((total + 255) / 256), 256, 0, stream>>>(src, dst, C, H);
    };
    auto convt = [&](const float* src, int wi, int bi, float* dst,
                     int Ci, int Co, int H, bool tanh_act) {
        dim3 g((unsigned)((Co + 63) / 64), (unsigned)((H * H) / 128), 4u);
        for (int dy = 0; dy < 2; ++dy)
            for (int dx = 0; dx < 2; ++dx) {
                if (tanh_act)
                    convt_wmma_k<2><<<g, 256, 0, stream>>>(src, fin(wi), fin(bi), dst, Ci, Co, H, H, dy, dx);
                else
                    convt_wmma_k<0><<<g, 256, 0, stream>>>(src, fin(wi), fin(bi), dst, Ci, Co, H, H, dy, dx);
            }
    };

    // ------------------ Encoder: 5 x (block -> maxpool) ------------------
    static const int eci[5] = {3, 64, 128, 256, 512};
    static const int eco[5] = {64, 128, 256, 512, 1024};
    const float* cur = fin(51);   // x
    float* pA = bufA;
    float* pB = bufB;
    int H = 512;
    for (int s = 0; s < 5; ++s) {
        conv3(cur, 27 + 4 * s, 28 + 4 * s, pA, eci[s], eco[s], H, true);
        conv3(pA, 29 + 4 * s, 30 + 4 * s, pB, eco[s], eco[s], H, true);
        pool(pB, pA, eco[s], H);
        H >>= 1;
        cur = pA;
        float* tmp = pA; pA = pB; pB = tmp;
    }
    // cur: [4,1024,16,16], H == 16; pA is the free buffer now.

    // ------------------ Vector quantization ------------------
    float* zE = pA;                                // z_e [4,64,16,16]
    conv1(cur, 50, 49, zE, 1024, 64, 16);          // pre: 1x1 1024->64
    emb_norm_k<<<4, 256, 0, stream>>>(fin(26), enorm);
    zero_k<<<1, 32, 0, stream>>>(lsum);
    vq_argmin_k<<<64, 256, 0, stream>>>(zE, fin(26), enorm, vidx);
    vq_gather_k<<<4, 256, 0, stream>>>(vidx, fin(26), zE, latent, lsum);
    qloss_k<<<1, 32, 0, stream>>>(lsum, qloss);

    // ------------------ Decoder ------------------
    conv1(latent, 48, 47, bufA, 64, 1024, 16);     // post: 1x1 64->1024
    static const int dci[5] = {1024, 512, 256, 128, 64};
    static const int dco[5] = {512, 256, 128, 64, 3};
    float* c = bufA;
    float* o = bufB;
    H = 16;
    for (int i = 0; i < 4; ++i) {
        convt(c, 16 + 2 * i, 17 + 2 * i, o, dci[i], dco[i], H, false);
        H <<= 1;
        conv3(o, 4 * i + 0, 4 * i + 1, c, dco[i], dco[i], H, true);
        conv3(c, 4 * i + 2, 4 * i + 3, o, dco[i], dco[i], H, true);
        float* tmp = c; c = o; o = tmp;
    }
    // c: [4,64,256,256]; final convT 64->3 with fused tanh -> recon [4,3,512,512]
    convt(c, 16 + 2 * 4, 17 + 2 * 4, recon, dci[4], dco[4], H, true);
}
